// WeightOnlyInt4Linear_21706764714506
// MI455X (gfx1250) — matrix operand order repaired and checked by reading, not compiled, not run on hardware
//
#include <hip/hip_runtime.h>

// ---- types ----------------------------------------------------------------
typedef __attribute__((ext_vector_type(16))) __bf16 v16bf;
typedef __attribute__((ext_vector_type(8)))  __bf16 v8bf;
typedef __attribute__((ext_vector_type(4)))  __bf16 v4bf;
typedef __attribute__((ext_vector_type(8)))  float  v8f;

// ---- problem constants (from reference: B=4,S=2048,K=4096,O=11008,G=128) --
#define M_TOT 8192
#define K_TOT 4096
#define N_TOT 11008
#define GRP   128

// ---- tiling ---------------------------------------------------------------
#define BM 128
#define BN 128
#define BK 32
#define KITERS (K_TOT / BK)          // 128
#define LDA 80                        // bytes/row: 32 bf16 (64B) + 16B pad -> conflict-free b128
#define LDB 80
#define A_BYTES (BM * LDA)            // 10240
#define B_BYTES (BN * LDB)            // 10240
#define STAGE_BYTES (A_BYTES + B_BYTES)

union FragU { v16bf v; v8bf h[2]; };

__global__ __launch_bounds__(256)
void WeightOnlyInt4Linear_wmma_kernel(const float* __restrict__ x,
                                      const int*   __restrict__ wq,
                                      const float* __restrict__ saz,
                                      float*       __restrict__ y)
{
    __shared__ __align__(16) unsigned char smem[2 * STAGE_BYTES];

    // ---- block swizzle: supergroups of 32 M-tiles sweep all N-tiles -------
    const int NT = N_TOT / BN;                 // 86
    const int GM = 32;
    const int per = GM * NT;
    const int bid = blockIdx.x;
    const int grp = bid / per;
    const int rem = bid - grp * per;
    const int mi  = grp * GM + (rem % GM);
    const int ni  = rem / GM;
    const int bm  = mi * BM;
    const int bn  = ni * BN;

    const int tid  = threadIdx.x;
    const int lane = tid & 31;
    const int wid  = tid >> 5;                 // 8 waves
    const int wm   = wid >> 2;                 // 0..1 -> 64 M rows
    const int wn   = wid & 3;                  // 0..3 -> 32 N cols
    const int hsel = lane >> 4;                // half-wave select
    const int l16  = lane & 15;

    // ---- loader geometry: 8 threads cover one 32-elem K row, 32 rows/pass -
    const int lrow = tid >> 3;                 // 0..31
    const int lk   = (tid & 7) * 4;            // k offset, 4 elems

    float4 ra[4];
    int4   rq[4];
    float2 rs[4];

    auto loadRegs = [&](int k0) {
        const float* xp = x + (size_t)(bm + lrow) * K_TOT + k0 + lk;
        #pragma unroll
        for (int p = 0; p < 4; ++p)
            ra[p] = *(const float4*)(xp + (size_t)p * 32 * K_TOT);
        const int* qp = wq + (size_t)(bn + lrow) * K_TOT + k0 + lk;
        #pragma unroll
        for (int p = 0; p < 4; ++p)
            rq[p] = *(const int4*)(qp + (size_t)p * 32 * K_TOT);
        const int g = k0 / GRP;                // one group per K-tile (32 | 128)
        const float* sp = saz + ((size_t)g * N_TOT + (bn + lrow)) * 2;
        #pragma unroll
        for (int p = 0; p < 4; ++p)
            rs[p] = *(const float2*)(sp + (size_t)p * 32 * 2);
    };

    auto storeRegs = [&](int st) {
        unsigned char* aB = smem + st * STAGE_BYTES;
        unsigned char* bB = aB + A_BYTES;
        #pragma unroll
        for (int p = 0; p < 4; ++p) {
            v4bf av;
            av[0] = (__bf16)ra[p].x; av[1] = (__bf16)ra[p].y;
            av[2] = (__bf16)ra[p].z; av[3] = (__bf16)ra[p].w;
            *(v4bf*)(aB + (p * 32 + lrow) * LDA + lk * 2) = av;

            const float s0 = rs[p].x, z0 = rs[p].y;
            v4bf bv;
            bv[0] = (__bf16)((float)(rq[p].x - 8) * s0 + z0);
            bv[1] = (__bf16)((float)(rq[p].y - 8) * s0 + z0);
            bv[2] = (__bf16)((float)(rq[p].z - 8) * s0 + z0);
            bv[3] = (__bf16)((float)(rq[p].w - 8) * s0 + z0);
            *(v4bf*)(bB + (p * 32 + lrow) * LDB + lk * 2) = bv;
        }
    };

    v8f c[4][2] = {};

    loadRegs(0);

    for (int kt = 0; kt < KITERS; ++kt) {
        const int st = kt & 1;
        storeRegs(st);
        __syncthreads();

        if (kt + 1 < KITERS) {
            loadRegs((kt + 1) * BK);
            if (kt + 2 < KITERS) {      // -> global_prefetch_b8
                __builtin_prefetch(wq + (size_t)(bn + lrow) * K_TOT + (kt + 2) * BK + lk, 0, 0);
                __builtin_prefetch(x  + (size_t)(bm + lrow) * K_TOT + (kt + 2) * BK + lk, 0, 0);
            }
        }

        const unsigned char* aB = smem + st * STAGE_BYTES;
        const unsigned char* bB = aB + A_BYTES;

        // A frags: 16x32 bf16; element e needs K = 16*(e>>3) + hsel*8 + (e&7)
        FragU a[4];
        #pragma unroll
        for (int i = 0; i < 4; ++i) {
            const unsigned char* p = aB + (wm * 64 + i * 16 + l16) * LDA + hsel * 16;
            a[i].h[0] = *(const v8bf*)p;          // K = hsel*8 .. +7
            a[i].h[1] = *(const v8bf*)(p + 32);   // K = 16+hsel*8 .. +7
        }
        // B frags: 32x16 bf16; element e needs K = 16*(e>>3) + hsel*8 + (e&7)
        FragU b[2];
        #pragma unroll
        for (int j = 0; j < 2; ++j) {
            const unsigned char* p = bB + (wn * 32 + j * 16 + l16) * LDB + hsel * 16;
            b[j].h[0] = *(const v8bf*)p;          // K = hsel*8 .. +7
            b[j].h[1] = *(const v8bf*)(p + 32);   // K = 16+hsel*8 .. +7
        }

        #pragma unroll
        for (int i = 0; i < 4; ++i)
            #pragma unroll
            for (int j = 0; j < 2; ++j)
                c[i][j] = __builtin_amdgcn_wmma_f32_16x16x32_bf16(
                    false, a[i].v, false, b[j].v, (short)0, c[i][j], false, false);

        __syncthreads();
    }

    // ---- epilogue: C/D 16x16 f32 layout: VGPR v -> M = v + hsel*8, N = l16
    // non-temporal stores keep the 344 MiB output out of L2 (preserve weights)
    #pragma unroll
    for (int i = 0; i < 4; ++i) {
        #pragma unroll
        for (int j = 0; j < 2; ++j) {
            const int col = bn + wn * 32 + j * 16 + l16;
            #pragma unroll
            for (int v = 0; v < 8; ++v) {
                const int row = bm + wm * 64 + i * 16 + hsel * 8 + v;
                __builtin_nontemporal_store(c[i][j][v], y + (size_t)row * N_TOT + col);
            }
        }
    }
}

extern "C" void kernel_launch(void* const* d_in, const int* in_sizes, int n_in,
                              void* d_out, int out_size, void* d_ws, size_t ws_size,
                              hipStream_t stream) {
    (void)in_sizes; (void)n_in; (void)out_size; (void)d_ws; (void)ws_size;
    const float* x   = (const float*)d_in[0];
    const int*   wq  = (const int*)d_in[1];
    const float* saz = (const float*)d_in[2];
    float*       y   = (float*)d_out;

    const int blocks = (M_TOT / BM) * (N_TOT / BN);   // 64 * 86 = 5504
    WeightOnlyInt4Linear_wmma_kernel<<<dim3(blocks), dim3(256), 0, stream>>>(x, wq, saz, y);
}
